// DoubleConv_27882927686272
// MI455X (gfx1250) — matrix operand order repaired and checked by reading, not compiled, not run on hardware
//
#include <hip/hip_runtime.h>
#include <cstdint>
#include <cstddef>

// ---------------------------------------------------------------------------
// Problem constants (from reference): N=4, Cin=64, Cmid=Cout=128, H=W=96, K=9
// ---------------------------------------------------------------------------
constexpr int Nn = 4, Hh = 96, Ww = 96;
constexpr int HW = Hh * Ww;            // 9216
constexpr int NPIX = Nn * HW;          // 36864
constexpr int C1 = 64, CM = 128, CO = 128, KK = 9;
constexpr int CK1 = KK * C1;           // 576
constexpr int CK2 = KK * CM;           // 1152

typedef __bf16 bf16x8  __attribute__((ext_vector_type(8)));
typedef __bf16 bf16x16 __attribute__((ext_vector_type(16)));
typedef float  f32x8   __attribute__((ext_vector_type(8)));
typedef int    i32x4   __attribute__((ext_vector_type(4)));

__device__ __forceinline__ uint16_t f2bf(float f) {
    uint32_t u = __builtin_bit_cast(uint32_t, f);
    u += 0x7FFFu + ((u >> 16) & 1u);          // round-to-nearest-even
    return (uint16_t)(u >> 16);
}

// ---------------------------------------------------------------------------
// Async global->LDS staging (gfx1250 GLOBAL_LOAD_ASYNC_TO_LDS_B128, ASYNCcnt)
// Builtin exists on this toolchain; param 0 is a non-const AS(1) int4*
// (diagnostic: "__vector_size__(4*sizeof(int)) int __device__ *").
// ---------------------------------------------------------------------------
#if __has_builtin(__builtin_amdgcn_global_load_async_to_lds_b128)
#define ATHENA_ASYNC_LDS 1
#else
#define ATHENA_ASYNC_LDS 0
#endif

__device__ __forceinline__ void stage16B(const uint16_t* gsrc, uint16_t* ldst) {
#if ATHENA_ASYNC_LDS
    __builtin_amdgcn_global_load_async_to_lds_b128(
        (__attribute__((address_space(1))) i32x4*)gsrc,
        (__attribute__((address_space(3))) i32x4*)ldst, 0, 0);
#else
    *(uint4*)ldst = *(const uint4*)gsrc;
#endif
}

__device__ __forceinline__ void stage_wait() {
#if ATHENA_ASYNC_LDS
#if __has_builtin(__builtin_amdgcn_s_wait_asynccnt)
    __builtin_amdgcn_s_wait_asynccnt(0);
#else
    asm volatile("s_wait_asynccnt 0x0" ::: "memory");
#endif
#endif
}

// ---------------------------------------------------------------------------
// NCHW (harness input) -> CNHW (internal layout: channel-major planes)
// ---------------------------------------------------------------------------
__global__ void k_transpose_in(const float* __restrict__ x, float* __restrict__ xt) {
    int tid = blockIdx.x * blockDim.x + threadIdx.x;
    if (tid >= C1 * NPIX) return;
    int c = tid / NPIX, pix = tid % NPIX;
    int n = pix / HW, hw = pix % HW;
    xt[tid] = x[((size_t)n * C1 + c) * HW + hw];
}

// ---------------------------------------------------------------------------
// Plain 3x3 conv (pad 1) producing 27 offset/mask channels. in is [C][N][H][W]
// ---------------------------------------------------------------------------
__global__ void k_offconv(const float* __restrict__ in, int C,
                          const float* __restrict__ offw,  // [27][C][3][3]
                          const float* __restrict__ offb,  // [27]
                          float* __restrict__ off) {       // [27][NPIX]
    int tid = blockIdx.x * blockDim.x + threadIdx.x;
    if (tid >= 27 * NPIX) return;
    int oc = tid / NPIX, pix = tid % NPIX;
    int n = pix / HW, hw = pix % HW;
    int h = hw / Ww, w = hw % Ww;
    float acc = offb[oc];
    for (int c = 0; c < C; ++c) {
        const float* pl  = in + ((size_t)c * Nn + n) * HW;
        const float* wpt = offw + ((size_t)oc * C + c) * 9;
        #pragma unroll
        for (int kh = 0; kh < 3; ++kh) {
            int ih = h + kh - 1;
            if (ih < 0 || ih >= Hh) continue;
            #pragma unroll
            for (int kw = 0; kw < 3; ++kw) {
                int iw = w + kw - 1;
                if (iw < 0 || iw >= Ww) continue;
                acc += wpt[kh * 3 + kw] * pl[ih * Ww + iw];
            }
        }
    }
    off[(size_t)oc * NPIX + pix] = acc;
}

// ---------------------------------------------------------------------------
// Repack conv weights (O,C,3,3) f32 -> Wp[o][k*C + c] bf16
// ---------------------------------------------------------------------------
__global__ void k_repack_w(const float* __restrict__ w, uint16_t* __restrict__ Wp,
                           int C, int total) {
    int tid = blockIdx.x * blockDim.x + threadIdx.x;
    if (tid >= total) return;
    int CK = C * KK;
    int o = tid / CK, r = tid % CK;
    int k = r / C, c = r % C;
    Wp[(size_t)o * CK + k * C + c] = f2bf(w[((size_t)o * C + c) * KK + k]);
}

// ---------------------------------------------------------------------------
// im2col with bilinear sampling + sigmoid mask modulation.
// One thread per (pix, k); corner weights shared across all C channels.
// col layout: [NPIX][K*C] (ck = k*C + c, contiguous in c) -> bf16
// ---------------------------------------------------------------------------
__global__ void k_sample(const float* __restrict__ in,   // [C][N][H][W]
                         const float* __restrict__ off,  // [27][NPIX]
                         uint16_t* __restrict__ col, int C) {
    int tid = blockIdx.x * blockDim.x + threadIdx.x;
    if (tid >= NPIX * KK) return;
    int k = tid / NPIX, pix = tid % NPIX;
    int n = pix / HW, hw = pix % HW;
    int h = hw / Ww, w = hw % Ww;

    float dy = off[(size_t)(2 * k) * NPIX + pix];
    float dx = off[(size_t)(2 * k + 1) * NPIX + pix];
    float mr = off[(size_t)(2 * KK + k) * NPIX + pix];
    float m  = 1.0f / (1.0f + expf(-mr));

    float py = dy + (float)(k / 3 - 1) + (float)h;
    float px = dx + (float)(k % 3 - 1) + (float)w;
    float y0f = floorf(py), x0f = floorf(px);
    float ly = py - y0f, lx = px - x0f;
    int y0 = (int)y0f, x0 = (int)x0f;
    int y1 = y0 + 1,  x1 = x0 + 1;

    float vy0 = (y0 >= 0 && y0 < Hh) ? 1.f : 0.f;
    float vy1 = (y1 >= 0 && y1 < Hh) ? 1.f : 0.f;
    float vx0 = (x0 >= 0 && x0 < Ww) ? 1.f : 0.f;
    float vx1 = (x1 >= 0 && x1 < Ww) ? 1.f : 0.f;

    float w00 = (1.f - ly) * (1.f - lx) * vy0 * vx0 * m;
    float w01 = (1.f - ly) * lx         * vy0 * vx1 * m;
    float w10 = ly         * (1.f - lx) * vy1 * vx0 * m;
    float w11 = ly         * lx         * vy1 * vx1 * m;

    int y0c = min(max(y0, 0), Hh - 1), y1c = min(max(y1, 0), Hh - 1);
    int x0c = min(max(x0, 0), Ww - 1), x1c = min(max(x1, 0), Ww - 1);

    int CK = C * KK;
    size_t colbase = (size_t)pix * CK + (size_t)k * C;
    for (int c = 0; c < C; ++c) {
        const float* pl = in + ((size_t)c * Nn + n) * HW;
        float v = w00 * pl[y0c * Ww + x0c] + w01 * pl[y0c * Ww + x1c]
                + w10 * pl[y1c * Ww + x0c] + w11 * pl[y1c * Ww + x1c];
        col[colbase + c] = f2bf(v);
    }
}

// ---------------------------------------------------------------------------
// WMMA GEMM: y[o][pix] = sum_ck Wp[o][ck] * col[pix][ck]
//
// Block = 8 waves = full o range (128) x 128 pixels (8 sub-tiles of 16).
// Per K-step (32): the 128px x 32k bf16 B tile (8 KB) is staged into LDS once
// via GLOBAL_LOAD_ASYNC_TO_LDS_B128 (double buffered, ASYNCcnt + barrier),
// each wave loads its A slice once from global and issues 8 WMMAs against
// LDS-resident B sub-tiles. Last K-step peeled -> branch-free staging.
// Operand k order: contiguous, k = 16*hi + s in both A and B (v_wmma_f32_16x16x32_bf16, wave32).
// No divergent control flow anywhere -> EXEC all ones for every WMMA.
// ---------------------------------------------------------------------------
__global__ void __launch_bounds__(256)
k_gemm_wmma(const uint16_t* __restrict__ Wp, const uint16_t* __restrict__ col,
            float* __restrict__ y, int CK) {
    __shared__ alignas(16) uint16_t Bbuf[2][128 * 32];   // 2 x 8 KB double buffer

    const int tid  = threadIdx.x;
    const int wave = tid >> 5;              // 0..7 -> o tile
    const int lane = tid & 31;
    const int lr   = lane & 15;             // row (A) / column (B,D)
    const int hi   = lane >> 4;             // half-wave select
    const int o0   = wave * 16;
    const int pix0 = blockIdx.x * 128;

    // cooperative B staging: thread copies 32 B (two b128) per K-step
    const int spx   = tid >> 1;             // 0..127 : local pixel
    const int skoff = (tid & 1) * 16;       // 0 or 16 : k offset (bf16 elems)
    const uint16_t* bsrc = col + (size_t)(pix0 + spx) * CK + skoff;
    const int sdst = spx * 32 + skoff;      // element offset inside a buffer

    const uint16_t* arow = Wp + (size_t)(o0 + lr) * CK + 16 * hi;

    f32x8 acc[8];
    #pragma unroll
    for (int t = 0; t < 8; ++t) acc[t] = (f32x8){0.f,0.f,0.f,0.f,0.f,0.f,0.f,0.f};

    const int steps = CK / 32;

    auto stage = [&](int s, int buf) {
        const uint16_t* src = bsrc + s * 32;
        stage16B(src,     &Bbuf[buf][sdst]);
        stage16B(src + 8, &Bbuf[buf][sdst + 8]);
    };

    auto compute = [&](int s) {
        const int k0 = s * 32;
        // A slice: elements 0..15 = K (k0 + 16*hi + 0..15)
        bf16x8 alo = *reinterpret_cast<const bf16x8*>(arow + k0);
        bf16x8 ahi = *reinterpret_cast<const bf16x8*>(arow + k0 + 8);
        bf16x16 a = __builtin_shufflevector(alo, ahi,
                        0, 1, 2, 3, 4, 5, 6, 7, 8, 9, 10, 11, 12, 13, 14, 15);
        __builtin_prefetch((const char*)(arow + k0 + 64), 0, 3);

        // B sub-tiles from LDS: column = 16t + lr, K = k0 + 16*hi + 0..15
        const uint16_t* brd = &Bbuf[s & 1][lr * 32 + 16 * hi];
        #pragma unroll
        for (int t = 0; t < 8; ++t) {
            bf16x16 b = *reinterpret_cast<const bf16x16*>(brd + t * 16 * 32);
            acc[t] = __builtin_amdgcn_wmma_f32_16x16x32_bf16(
                         false, a, false, b, (short)0, acc[t], false, false);
        }
    };

    // prologue: stage K-step 0 into buffer 0
    stage(0, 0);
    stage_wait();
    __syncthreads();

    for (int s = 0; s < steps - 1; ++s) {
        stage(s + 1, (s + 1) & 1);          // branch-free in-loop staging
        compute(s);
        stage_wait();
        __syncthreads();
    }
    compute(steps - 1);                     // peeled final K-step

    // D: VGPR j holds row o0 + 8*hi + j, column pix0 + 16*t + lr
    const size_t orow = (size_t)(o0 + 8 * hi);
    #pragma unroll
    for (int j = 0; j < 8; ++j) {
        float* yr = y + (orow + j) * NPIX + pix0 + lr;
        #pragma unroll
        for (int t = 0; t < 8; ++t)
            yr[16 * t] = acc[t][j];
    }
}

// ---------------------------------------------------------------------------
// Per-channel batch stats over NPIX samples: stats[c]=mean, stats[128+c]=rstd
// ---------------------------------------------------------------------------
__global__ void k_bnstats(const float* __restrict__ y, float* __restrict__ stats) {
    __shared__ float s1[256], s2[256];
    int c = blockIdx.x;
    const float* row = y + (size_t)c * NPIX;
    float sum = 0.f, sq = 0.f;
    for (int i = threadIdx.x; i < NPIX; i += 256) {
        float v = row[i];
        sum += v; sq += v * v;
    }
    s1[threadIdx.x] = sum; s2[threadIdx.x] = sq;
    __syncthreads();
    for (int s = 128; s > 0; s >>= 1) {
        if (threadIdx.x < s) {
            s1[threadIdx.x] += s1[threadIdx.x + s];
            s2[threadIdx.x] += s2[threadIdx.x + s];
        }
        __syncthreads();
    }
    if (threadIdx.x == 0) {
        float mean = s1[0] / (float)NPIX;
        float var  = s2[0] / (float)NPIX - mean * mean;
        stats[c]       = mean;
        stats[128 + c] = rsqrtf(var + 1e-5f);
    }
}

// BN+ReLU, output kept in internal CNHW layout (feeds layer 2)
__global__ void k_bnrelu_cnhw(const float* __restrict__ y, const float* __restrict__ stats,
                              const float* __restrict__ g, const float* __restrict__ b,
                              float* __restrict__ out) {
    int tid = blockIdx.x * blockDim.x + threadIdx.x;
    if (tid >= 128 * NPIX) return;
    int c = tid / NPIX;
    float v = (y[tid] - stats[c]) * stats[128 + c] * g[c] + b[c];
    out[tid] = v > 0.f ? v : 0.f;
}

// BN+ReLU, output written in harness NCHW layout
__global__ void k_bnrelu_nchw(const float* __restrict__ y, const float* __restrict__ stats,
                              const float* __restrict__ g, const float* __restrict__ b,
                              float* __restrict__ out) {
    int tid = blockIdx.x * blockDim.x + threadIdx.x;
    if (tid >= 128 * NPIX) return;
    int c = tid / NPIX, pix = tid % NPIX;
    int n = pix / HW, hw = pix % HW;
    float v = (y[tid] - stats[c]) * stats[128 + c] * g[c] + b[c];
    out[((size_t)n * 128 + c) * HW + hw] = v > 0.f ? v : 0.f;
}

// ---------------------------------------------------------------------------
// Workspace layout (bytes, 256-aligned regions; total ~136.4 MB)
// ---------------------------------------------------------------------------
constexpr size_t OFF_XT    = 0;                                  // 64 *NPIX*4  =  9,437,184
constexpr size_t OFF_H     = OFF_XT    + (size_t)C1 * NPIX * 4;  // 128*NPIX*4  = 18,874,368
constexpr size_t OFF_Y     = OFF_H     + (size_t)CM * NPIX * 4;  // 128*NPIX*4  = 18,874,368
constexpr size_t OFF_OFF   = OFF_Y     + (size_t)CO * NPIX * 4;  // 27 *NPIX*4  =  3,981,312
constexpr size_t OFF_STATS = OFF_OFF   + (size_t)27 * NPIX * 4;  // 256*4       =      1,024
constexpr size_t OFF_WP    = OFF_STATS + 1024;                   // 128*1152*2  =    294,912
constexpr size_t OFF_COL   = OFF_WP    + (size_t)CO * CK2 * 2;   // NPIX*1152*2 = 84,934,656

extern "C" void kernel_launch(void* const* d_in, const int* in_sizes, int n_in,
                              void* d_out, int out_size, void* d_ws, size_t ws_size,
                              hipStream_t stream) {
    const float* x      = (const float*)d_in[0];
    const float* w1     = (const float*)d_in[1];
    const float* off_w1 = (const float*)d_in[2];
    const float* off_b1 = (const float*)d_in[3];
    const float* gamma1 = (const float*)d_in[4];
    const float* beta1  = (const float*)d_in[5];
    const float* w2     = (const float*)d_in[6];
    const float* off_w2 = (const float*)d_in[7];
    const float* off_b2 = (const float*)d_in[8];
    const float* gamma2 = (const float*)d_in[9];
    const float* beta2  = (const float*)d_in[10];
    float* out = (float*)d_out;

    char* ws = (char*)d_ws;
    float*    xt    = (float*)   (ws + OFF_XT);
    float*    h     = (float*)   (ws + OFF_H);
    float*    y     = (float*)   (ws + OFF_Y);
    float*    off   = (float*)   (ws + OFF_OFF);
    float*    stats = (float*)   (ws + OFF_STATS);
    uint16_t* Wp    = (uint16_t*)(ws + OFF_WP);
    uint16_t* col   = (uint16_t*)(ws + OFF_COL);

    const dim3 blk(256);

    // ---- Layer 1 (Cin=64 -> Cmid=128) ----
    k_transpose_in<<<dim3((C1 * NPIX) / 256), blk, 0, stream>>>(x, xt);
    k_offconv<<<dim3((27 * NPIX) / 256), blk, 0, stream>>>(xt, C1, off_w1, off_b1, off);
    k_repack_w<<<dim3((CM * CK1) / 256), blk, 0, stream>>>(w1, Wp, C1, CM * CK1);
    k_sample<<<dim3((NPIX * KK + 255) / 256), blk, 0, stream>>>(xt, off, col, C1);
    k_gemm_wmma<<<dim3(NPIX / 128), blk, 0, stream>>>(Wp, col, y, CK1);
    k_bnstats<<<dim3(CM), blk, 0, stream>>>(y, stats);
    k_bnrelu_cnhw<<<dim3((CM * NPIX) / 256), blk, 0, stream>>>(y, stats, gamma1, beta1, h);

    // ---- Layer 2 (Cmid=128 -> Cout=128) ----
    k_offconv<<<dim3((27 * NPIX) / 256), blk, 0, stream>>>(h, CM, off_w2, off_b2, off);
    k_repack_w<<<dim3((CO * CK2) / 256), blk, 0, stream>>>(w2, Wp, CM, CO * CK2);
    k_sample<<<dim3((NPIX * KK + 255) / 256), blk, 0, stream>>>(h, off, col, CM);
    k_gemm_wmma<<<dim3(NPIX / 128), blk, 0, stream>>>(Wp, col, y, CK2);
    k_bnstats<<<dim3(CO), blk, 0, stream>>>(y, stats);
    k_bnrelu_nchw<<<dim3((CO * NPIX) / 256), blk, 0, stream>>>(y, stats, gamma2, beta2, out);
}
